// Attention_14474039788120
// MI455X (gfx1250) — compile-verified, matrix operand order checked by reading, not run on hardware
//
#include <hip/hip_runtime.h>
#include <math.h>

typedef __attribute__((ext_vector_type(16))) _Float16 v16h;
typedef __attribute__((ext_vector_type(8)))  _Float16 v8h;
typedef __attribute__((ext_vector_type(4)))  _Float16 v4h;
typedef __attribute__((ext_vector_type(8)))  float    v8f;

#define B_  2
#define S_  2048
#define D_  1024
#define H_  16
#define HD_ 64

__device__ __forceinline__ v16h cat8(v8h lo, v8h hi) {
    return __builtin_shufflevector(lo, hi, 0,1,2,3,4,5,6,7,8,9,10,11,12,13,14,15);
}

// ---------------- elementwise converts ----------------

__global__ __launch_bounds__(256) void cvt_f32_f16(const float* __restrict__ src,
                                                   _Float16* __restrict__ dst, int n) {
    int i = (blockIdx.x * 256 + threadIdx.x) * 4;
    if (i < n) {
        float4 v = *(const float4*)(src + i);
        v4h o = { (_Float16)v.x, (_Float16)v.y, (_Float16)v.z, (_Float16)v.w };
        *(v4h*)(dst + i) = o;
    }
}

// W[k][n] f32 -> Wt[n][k] f16
__global__ __launch_bounds__(256) void cvt_transpose(const float* __restrict__ W,
                                                     _Float16* __restrict__ Wt,
                                                     int K, int N) {
    int idx = blockIdx.x * 256 + threadIdx.x;   // over K*N, coalesced read
    int k = idx / N, n = idx % N;
    Wt[(size_t)n * K + k] = (_Float16)W[idx];
}

// ---------------- WMMA GEMM: C[M,N] = X[M,K] * Wt[N,K]^T + bias ----------------
// Block tile 128x128, 8 waves (4x2), wave tile 32x64 (2x4 WMMA tiles), BK=32,
// register-prefetch software pipeline on the K loop.
// mode 0: dst f32 [M][D_]
// mode 1: dst f16 [B,H,S,HD]  (val*scale)   (Q with scale=0.125, K with 1.0)
// mode 2: dst f16 [B,H,HD,S]  (V transposed for attention)

__global__ __launch_bounds__(256)
void gemm_wmma(const _Float16* __restrict__ X, const _Float16* __restrict__ Wt,
               const float* __restrict__ bias, void* __restrict__ dstv,
               int Kdim, int mode, float scale) {
    __shared__ __align__(16) _Float16 As[128][32];
    __shared__ __align__(16) _Float16 Bs[128][32];

    const int t    = threadIdx.x;
    const int lane = t & 31;
    const int wave = t >> 5;          // 0..7
    const int wm   = wave >> 1;       // 0..3 -> 32-row slab
    const int wn   = wave & 1;        // 0..1 -> 64-col slab
    const int l15  = lane & 15;
    const int hs   = lane >> 4;       // half-wave select
    const int kb   = hs * 8;

    const int blockM = blockIdx.y * 128;
    const int blockN = blockIdx.x * 128;

    // staging coordinates (two 16B chunks per thread per tile)
    const int srow0 = t >> 2;                 // chunk 0 row (0..63)
    const int srow1 = (t + 256) >> 2;         // chunk 1 row (64..127)
    const int sc8   = (t & 3) * 8;            // col offset in halves

    v8f acc[2][4] = {};
    v8h regA[2], regB[2];

    // prologue: prefetch K-slice 0 into registers
    regA[0] = *(const v8h*)(X  + (size_t)(blockM + srow0) * Kdim + sc8);
    regA[1] = *(const v8h*)(X  + (size_t)(blockM + srow1) * Kdim + sc8);
    regB[0] = *(const v8h*)(Wt + (size_t)(blockN + srow0) * Kdim + sc8);
    regB[1] = *(const v8h*)(Wt + (size_t)(blockN + srow1) * Kdim + sc8);

    for (int k0 = 0; k0 < Kdim; k0 += 32) {
        // commit prefetched slice to LDS
        *(v8h*)&As[srow0][sc8] = regA[0];
        *(v8h*)&As[srow1][sc8] = regA[1];
        *(v8h*)&Bs[srow0][sc8] = regB[0];
        *(v8h*)&Bs[srow1][sc8] = regB[1];
        __syncthreads();

        // issue global loads for next slice (overlaps with WMMA chain)
        if (k0 + 32 < Kdim) {
            int kn = k0 + 32;
            regA[0] = *(const v8h*)(X  + (size_t)(blockM + srow0) * Kdim + kn + sc8);
            regA[1] = *(const v8h*)(X  + (size_t)(blockM + srow1) * Kdim + kn + sc8);
            regB[0] = *(const v8h*)(Wt + (size_t)(blockN + srow0) * Kdim + kn + sc8);
            regB[1] = *(const v8h*)(Wt + (size_t)(blockN + srow1) * Kdim + kn + sc8);
        }

        v16h a[2], b[4];
        #pragma unroll
        for (int mi = 0; mi < 2; ++mi) {
            int r = wm * 32 + mi * 16 + l15;
            a[mi] = cat8(*(const v8h*)&As[r][kb], *(const v8h*)&As[r][kb + 16]);
        }
        #pragma unroll
        for (int ni = 0; ni < 4; ++ni) {
            int r = wn * 64 + ni * 16 + l15;
            b[ni] = cat8(*(const v8h*)&Bs[r][kb], *(const v8h*)&Bs[r][kb + 16]);
        }
        #pragma unroll
        for (int mi = 0; mi < 2; ++mi)
            #pragma unroll
            for (int ni = 0; ni < 4; ++ni)
                acc[mi][ni] = __builtin_amdgcn_wmma_f32_16x16x32_f16(
                    false, a[mi], false, b[ni], (short)0, acc[mi][ni], false, false);
        __syncthreads();
    }

    // epilogue: C layout lane=column N, VGPR v = row (+8 for upper half-wave)
    #pragma unroll
    for (int mi = 0; mi < 2; ++mi)
        #pragma unroll
        for (int ni = 0; ni < 4; ++ni) {
            int n = blockN + wn * 64 + ni * 16 + l15;
            float bv = bias[n];
            #pragma unroll
            for (int v = 0; v < 8; ++v) {
                int m = blockM + wm * 32 + mi * 16 + v + hs * 8;
                float val = acc[mi][ni][v] + bv;
                if (mode == 0) {
                    ((float*)dstv)[(size_t)m * D_ + n] = val;
                } else {
                    int b  = m >> 11, s  = m & (S_ - 1);
                    int h  = n >> 6,  hd = n & (HD_ - 1);
                    _Float16 hv = (_Float16)(val * scale);
                    if (mode == 1)
                        ((_Float16*)dstv)[(((size_t)(b * H_ + h)) * S_ + s) * HD_ + hd] = hv;
                    else
                        ((_Float16*)dstv)[(((size_t)(b * H_ + h)) * HD_ + hd) * S_ + s] = hv;
                }
            }
        }
}

// ---------------- flash attention ----------------
// Q,K: f16 [B,H,S,HD] (Q pre-scaled by 1/sqrt(HD)); Vt: f16 [B,H,HD,S]
// Ctx out: f16 [B,S,D]  (D = H*HD)
// block = 256 thr (8 waves); each wave owns a 16-query tile; block covers 128
// queries of one (b,h); all waves share LDS-staged 64-key K/V tiles.

__global__ __launch_bounds__(256)
void flash_attn(const _Float16* __restrict__ Q, const _Float16* __restrict__ Kt,
                const _Float16* __restrict__ Vt, _Float16* __restrict__ Ctx) {
    __shared__ __align__(16) _Float16 Ks[64][64];      // keys x hd      (8 KB)
    __shared__ __align__(16) _Float16 Vs[64][64];      // hd x keys      (8 KB)
    __shared__ __align__(16) _Float16 Ps[8][16][64];   // per-wave P     (16 KB)

    const int t    = threadIdx.x;
    const int lane = t & 31;
    const int wave = t >> 5;
    const int l15  = lane & 15;
    const int hs   = lane >> 4;
    const int kb   = hs * 8;

    const int bh     = blockIdx.x >> 4;     // b*H + h
    const int qchunk = blockIdx.x & 15;
    const int q0     = qchunk * 128 + wave * 16;

    const _Float16* Qg = Q  + (size_t)bh * S_ * HD_;
    const _Float16* Kg = Kt + (size_t)bh * S_ * HD_;
    const _Float16* Vg = Vt + (size_t)bh * HD_ * S_;

    // Q fragments (16 x 64 as two K=32 A-fragments), loaded once
    v16h qf[2];
    #pragma unroll
    for (int f = 0; f < 2; ++f) {
        const _Float16* p = Qg + (size_t)(q0 + l15) * HD_ + f * 32 + kb;
        qf[f] = cat8(*(const v8h*)p, *(const v8h*)(p + 16));
    }

    float mrow[8], lrow[8];
    #pragma unroll
    for (int v = 0; v < 8; ++v) { mrow[v] = -1e30f; lrow[v] = 0.0f; }
    v8f acc[4] = {};

    for (int j = 0; j < S_; j += 64) {
        // stage K tile (64x64) and V tile (64x64): two 16B chunks per thread each
        #pragma unroll
        for (int i = 0; i < 2; ++i) {
            int idx = t + i * 256;            // 0..511
            int krow = idx >> 3, kc8 = (idx & 7) * 8;
            *(v8h*)&Ks[krow][kc8] = *(const v8h*)(Kg + (size_t)(j + krow) * HD_ + kc8);
            *(v8h*)&Vs[krow][kc8] = *(const v8h*)(Vg + (size_t)krow * S_ + j + kc8);
        }
        if (j + 64 < S_) {
            __builtin_prefetch((const void*)(Kg + (size_t)(j + 64) * HD_), 0, 1);
            __builtin_prefetch((const void*)(Vg + j + 64), 0, 1);
        }
        __syncthreads();

        // scores: 16 queries x 64 keys = four 16x16 C tiles, K-dim(hd)=64 via 2 WMMAs
        v8f sc[4];
        #pragma unroll
        for (int nt = 0; nt < 4; ++nt) {
            v8f c = {};
            int r = nt * 16 + l15;
            #pragma unroll
            for (int f = 0; f < 2; ++f) {
                v16h bfrag = cat8(*(const v8h*)&Ks[r][f * 32 + kb],
                                  *(const v8h*)&Ks[r][f * 32 + kb + 16]);
                c = __builtin_amdgcn_wmma_f32_16x16x32_f16(
                        false, qf[f], false, bfrag, (short)0, c, false, false);
            }
            sc[nt] = c;
        }

        // online softmax; row = one VGPR across a 16-lane half -> shfl_xor reduce
        #pragma unroll
        for (int v = 0; v < 8; ++v) {
            float mx = fmaxf(fmaxf(sc[0][v], sc[1][v]), fmaxf(sc[2][v], sc[3][v]));
            mx = fmaxf(mx, __shfl_xor(mx, 1, 32));
            mx = fmaxf(mx, __shfl_xor(mx, 2, 32));
            mx = fmaxf(mx, __shfl_xor(mx, 4, 32));
            mx = fmaxf(mx, __shfl_xor(mx, 8, 32));
            float mnew  = fmaxf(mrow[v], mx);
            float alpha = __expf(mrow[v] - mnew);
            float p[4], rs = 0.0f;
            #pragma unroll
            for (int nt = 0; nt < 4; ++nt) { p[nt] = __expf(sc[nt][v] - mnew); rs += p[nt]; }
            rs += __shfl_xor(rs, 1, 32);
            rs += __shfl_xor(rs, 2, 32);
            rs += __shfl_xor(rs, 4, 32);
            rs += __shfl_xor(rs, 8, 32);
            lrow[v] = lrow[v] * alpha + rs;
            mrow[v] = mnew;
            #pragma unroll
            for (int tt = 0; tt < 4; ++tt) acc[tt][v] *= alpha;
            int prow = v + hs * 8;
            #pragma unroll
            for (int nt = 0; nt < 4; ++nt)
                Ps[wave][prow][nt * 16 + l15] = (_Float16)p[nt];
        }
        asm volatile("s_wait_dscnt 0" ::: "memory");  // wave-local P LDS RAW

        // ctx += P(16x64) x V(64x64): two A-frags, 4 hd tiles x 2 key-halves
        v16h pf[2];
        #pragma unroll
        for (int f = 0; f < 2; ++f)
            pf[f] = cat8(*(const v8h*)&Ps[wave][l15][f * 32 + kb],
                         *(const v8h*)&Ps[wave][l15][f * 32 + kb + 16]);
        #pragma unroll
        for (int tt = 0; tt < 4; ++tt) {
            int hd = tt * 16 + l15;
            #pragma unroll
            for (int f = 0; f < 2; ++f) {
                v16h vf = cat8(*(const v8h*)&Vs[hd][f * 32 + kb],
                               *(const v8h*)&Vs[hd][f * 32 + kb + 16]);
                acc[tt] = __builtin_amdgcn_wmma_f32_16x16x32_f16(
                              false, pf[f], false, vf, (short)0, acc[tt], false, false);
            }
        }
        __syncthreads();   // protect Ks/Vs for next stage
    }

    // epilogue: ctx / l, store f16 [b, s, h*64+hd]
    const int bb = bh >> 4, hh = bh & 15;
    #pragma unroll
    for (int tt = 0; tt < 4; ++tt) {
        int hd = tt * 16 + l15;
        #pragma unroll
        for (int v = 0; v < 8; ++v) {
            int s = q0 + v + hs * 8;
            float val = acc[tt][v] / lrow[v];
            Ctx[((size_t)(bb * S_ + s)) * D_ + hh * HD_ + hd] = (_Float16)val;
        }
    }
}

// ---------------- host launch ----------------

extern "C" void kernel_launch(void* const* d_in, const int* in_sizes, int n_in,
                              void* d_out, int out_size, void* d_ws, size_t ws_size,
                              hipStream_t stream) {
    (void)in_sizes; (void)n_in; (void)out_size; (void)ws_size;
    const float* hs = (const float*)d_in[0];
    const float* Wq = (const float*)d_in[1];
    const float* bq = (const float*)d_in[2];
    const float* Wk = (const float*)d_in[3];
    const float* bk = (const float*)d_in[4];
    const float* Wv = (const float*)d_in[5];
    const float* bv = (const float*)d_in[6];
    const float* Wo = (const float*)d_in[7];
    const float* bo = (const float*)d_in[8];

    char* ws = (char*)d_ws;
    const size_t MB = 1u << 20;
    _Float16* Xh   = (_Float16*)(ws + 0);        //  8 MB: hidden f16 [4096][1024]
    _Float16* WqT  = (_Float16*)(ws + 8  * MB);  //  2 MB each, transposed f16
    _Float16* WkT  = (_Float16*)(ws + 10 * MB);
    _Float16* WvT  = (_Float16*)(ws + 12 * MB);
    _Float16* WoT  = (_Float16*)(ws + 14 * MB);
    _Float16* Qh   = (_Float16*)(ws + 16 * MB);  //  8 MB [B,H,S,HD]
    _Float16* Kh   = (_Float16*)(ws + 24 * MB);  //  8 MB [B,H,S,HD]
    _Float16* Vth  = (_Float16*)(ws + 32 * MB);  //  8 MB [B,H,HD,S]
    _Float16* Ctx  = (_Float16*)(ws + 40 * MB);  //  8 MB [B,S,D]

    const int Melems = B_ * S_ * D_;             // 4,194,304
    cvt_f32_f16 <<<Melems / (256 * 4), 256, 0, stream>>>(hs, Xh, Melems);
    cvt_transpose<<<(D_ * D_) / 256, 256, 0, stream>>>(Wq, WqT, D_, D_);
    cvt_transpose<<<(D_ * D_) / 256, 256, 0, stream>>>(Wk, WkT, D_, D_);
    cvt_transpose<<<(D_ * D_) / 256, 256, 0, stream>>>(Wv, WvT, D_, D_);
    cvt_transpose<<<(D_ * D_) / 256, 256, 0, stream>>>(Wo, WoT, D_, D_);

    dim3 gg(D_ / 128, (B_ * S_) / 128);          // (8, 32)
    gemm_wmma<<<gg, 256, 0, stream>>>(Xh, WqT, bq, Qh,  D_, 1, 0.125f); // Q*1/sqrt(64)
    gemm_wmma<<<gg, 256, 0, stream>>>(Xh, WkT, bk, Kh,  D_, 1, 1.0f);
    gemm_wmma<<<gg, 256, 0, stream>>>(Xh, WvT, bv, Vth, D_, 2, 1.0f);

    flash_attn<<<B_ * H_ * (S_ / 128), 256, 0, stream>>>(Qh, Kh, Vth, Ctx);

    gemm_wmma<<<gg, 256, 0, stream>>>(Ctx, WoT, bo, d_out, D_, 0, 1.0f);
}
